// TPllamaAttention_67095979098385
// MI455X (gfx1250) — matrix operand order repaired and checked by reading, not compiled, not run on hardware
//
#include <hip/hip_runtime.h>
#include <hip/hip_bf16.h>

// ---------------------------------------------------------------------------
// Llama attention layer for MI455X (gfx1250), bf16 WMMA pipeline.
// B=1, S=2048, E=4096, H=32, D=128.
// ---------------------------------------------------------------------------

#define S_LEN 2048
#define E_DIM 4096
#define H_NUM 32
#define D_DIM 128
#define HD    (H_NUM * D_DIM)   // 4096

typedef __attribute__((ext_vector_type(16))) __bf16 v16bf;
typedef __attribute__((ext_vector_type(8)))  float  v8f;

struct U8x4 { unsigned int u[8]; };   // 32B, same size as v16bf

static __device__ __forceinline__ unsigned short f2bf(float f) {
  __bf16 h = (__bf16)f;
  return __builtin_bit_cast(unsigned short, h);
}
static __device__ __forceinline__ float bf2f(unsigned short u) {
  __bf16 h = __builtin_bit_cast(__bf16, u);
  return (float)h;
}

// CDNA5 async global->LDS copy (no VGPR round trip, tracked by ASYNCcnt).
// Generic LDS pointer low 32 bits == wave-relative LDS byte offset (ISA 10.2).
static __device__ __forceinline__ void async_copy_b128(const void* gptr, void* lptr) {
  unsigned lds_off = (unsigned)(unsigned long long)lptr;
  asm volatile("global_load_async_to_lds_b128 %0, %1, off"
               :: "v"(lds_off), "v"(gptr)
               : "memory");
}
static __device__ __forceinline__ void wait_async() {
  asm volatile("s_wait_asynccnt 0" ::: "memory");
}

// A-fragment K-offset for V_WMMA_F32_16X16X32_BF16 (16-bit A, 16x32):
// VGPR v (pair of halves), lanes 0-15: K = {2v, 2v+1}
// lanes 16-31: +16 (contiguous, matches B fragments).
static __device__ __forceinline__ int a_k0(int v, int lane) {
  int k = 2 * v;
  return k + ((lane & 16) ? 16 : 0);
}

static __device__ __forceinline__ float redmax16(float v) {
#pragma unroll
  for (int off = 1; off < 16; off <<= 1) v = fmaxf(v, __shfl_xor(v, off, 16));
  return v;
}
static __device__ __forceinline__ float redsum16(float v) {
#pragma unroll
  for (int off = 1; off < 16; off <<= 1) v += __shfl_xor(v, off, 16);
  return v;
}

// ---------------------------------------------------------------------------
// fp32 -> bf16 elementwise convert (n must be a multiple of 4)
// ---------------------------------------------------------------------------
__global__ __launch_bounds__(256) void cvt_f32_bf16(const float* __restrict__ src,
                                                    unsigned short* __restrict__ dst,
                                                    int n4) {
  int i = blockIdx.x * blockDim.x + threadIdx.x;
  if (i >= n4) return;
  const float4 f = ((const float4*)src)[i];
  uint2 o;
  o.x = (unsigned)f2bf(f.x) | ((unsigned)f2bf(f.y) << 16);
  o.y = (unsigned)f2bf(f.z) | ((unsigned)f2bf(f.w) << 16);
  ((uint2*)dst)[i] = o;
}

// ---------------------------------------------------------------------------
// RoPE in-place on bf16 [S][H][D], interleaved pairs (even/odd), pos = s.
// ---------------------------------------------------------------------------
__global__ __launch_bounds__(256) void rope_bf16(unsigned short* __restrict__ buf) {
  int idx = blockIdx.x * blockDim.x + threadIdx.x;     // over S*H*(D/2)
  const int half = D_DIM / 2;
  if (idx >= S_LEN * H_NUM * half) return;
  int i = idx % half;
  int rem = idx / half;
  int h = rem % H_NUM;
  int s = rem / H_NUM;
  float freq = __powf(10000.0f, -((float)i) / (float)half);
  float ang = (float)s * freq;
  float sn, cs;
  __sincosf(ang, &sn, &cs);
  size_t off = ((size_t)s * H_NUM + h) * D_DIM + 2 * i;
  unsigned int u = *(unsigned int*)&buf[off];
  float x1 = bf2f((unsigned short)(u & 0xffff));
  float x2 = bf2f((unsigned short)(u >> 16));
  float r1 = x1 * cs - x2 * sn;
  float r2 = x1 * sn + x2 * cs;
  *(unsigned int*)&buf[off] = (unsigned)f2bf(r1) | ((unsigned)f2bf(r2) << 16);
}

// ---------------------------------------------------------------------------
// Tiled WMMA GEMM: C[M,N] = A[M,K] * B[K,N], A/B bf16 row-major.
// Block tile 64(M) x 128(N), 256 threads = 8 waves (4 x 2), wave = 16 x 64.
// A tile staged via GLOBAL_LOAD_ASYNC_TO_LDS_B128; B tile transposed into LDS
// with pair-packed ds_store_b32.
// ---------------------------------------------------------------------------
template <bool OUT_BF16>
__global__ __launch_bounds__(256) void gemm_wmma_bf16(const unsigned short* __restrict__ A,
                                                      const unsigned short* __restrict__ B,
                                                      void* __restrict__ Cout,
                                                      int M, int N, int K) {
  __shared__ unsigned short lds_a[64][32];     // [m][k]      4 KB
  __shared__ unsigned short lds_bt[128][32];   // [n][k] (transposed)  8 KB

  const int t = threadIdx.x;
  const int lane = t & 31;
  const int wid = t >> 5;
  const int wave_m = wid >> 1;        // 0..3
  const int wave_n = wid & 1;         // 0..1
  const int mb = blockIdx.y * 64;
  const int nb = blockIdx.x * 128;

  // staging coordinates (constant across K loop)
  const int a_row = t >> 2;           // 0..63
  const int a_col = (t & 3) * 8;      // 0,8,16,24
  const int b_kp = (t >> 4) * 2;      // 0,2,..,30 (even k row)
  const int b_n8 = (t & 15) * 8;      // 0..120

  v8f acc[4] = {};                    // 4 N-subtiles of 16x16

  for (int kc = 0; kc < K; kc += 32) {
    // ---- async stage A tile: 64x32 bf16, one b128 per thread
    async_copy_b128(&A[(size_t)(mb + a_row) * K + kc + a_col], &lds_a[a_row][a_col]);

    // ---- stage B tile transposed: two k-rows pair-packed -> ds_store_b32
    {
      uint4 g0 = *(const uint4*)&B[(size_t)(kc + b_kp) * N + nb + b_n8];
      uint4 g1 = *(const uint4*)&B[(size_t)(kc + b_kp + 1) * N + nb + b_n8];
      unsigned int w0[4] = {g0.x, g0.y, g0.z, g0.w};
      unsigned int w1[4] = {g1.x, g1.y, g1.z, g1.w};
#pragma unroll
      for (int e = 0; e < 8; ++e) {
        unsigned lo = (w0[e >> 1] >> ((e & 1) * 16)) & 0xffffu;
        unsigned hi = (w1[e >> 1] >> ((e & 1) * 16)) & 0xffffu;
        *(unsigned int*)&lds_bt[b_n8 + e][b_kp] = lo | (hi << 16);
      }
    }

    // ---- prefetch next K-step tiles into cache while we compute
    if (kc + 32 < K) {
      __builtin_prefetch(&A[(size_t)(mb + a_row) * K + kc + 32 + a_col], 0, 3);
      __builtin_prefetch(&B[(size_t)(kc + 32 + b_kp) * N + nb + b_n8], 0, 3);
    }

    wait_async();
    __syncthreads();

    // ---- A fragment (16x32), contiguous k = 16h + s (matches B)
    U8x4 au;
#pragma unroll
    for (int v = 0; v < 8; ++v)
      au.u[v] = *(const unsigned int*)&lds_a[wave_m * 16 + (lane & 15)][a_k0(v, lane)];
    v16bf afrag = __builtin_bit_cast(v16bf, au);

    // ---- 4 B fragments + WMMA
#pragma unroll
    for (int sub = 0; sub < 4; ++sub) {
      int nl = wave_n * 64 + sub * 16 + (lane & 15);
      int koff = (lane & 16) ? 16 : 0;
      v16bf bfrag = *(const v16bf*)&lds_bt[nl][koff];
      acc[sub] = __builtin_amdgcn_wmma_f32_16x16x32_bf16(
          false, afrag, false, bfrag, (short)0, acc[sub], false, false);
    }
    __syncthreads();
  }

  // ---- store C: VGPR r -> row r (lanes 0-15) / r+8 (lanes 16-31), col = lane%16
#pragma unroll
  for (int sub = 0; sub < 4; ++sub) {
#pragma unroll
    for (int r = 0; r < 8; ++r) {
      int m = mb + wave_m * 16 + r + ((lane & 16) ? 8 : 0);
      int n = nb + wave_n * 64 + sub * 16 + (lane & 15);
      float val = acc[sub][r];
      if (OUT_BF16)
        ((unsigned short*)Cout)[(size_t)m * N + n] = f2bf(val);
      else
        ((float*)Cout)[(size_t)m * N + n] = val;
    }
  }
}

// ---------------------------------------------------------------------------
// Flash attention, causal. Block = 128 threads = 4 waves; each wave owns 16
// query rows of one head; K/V tiles (32 keys) staged in LDS per iteration.
// K tile staged via async global->LDS; V tile transposed with pair-packed
// ds_store_b32. Q/K already RoPE'd, bf16 [S][H][D]. Output bf16 [S][H*D].
// ---------------------------------------------------------------------------
__global__ __launch_bounds__(128) void attn_flash_wmma(const unsigned short* __restrict__ Qb,
                                                       const unsigned short* __restrict__ Kb,
                                                       const unsigned short* __restrict__ Vb,
                                                       unsigned short* __restrict__ Ob) {
  __shared__ unsigned short lds_k[32][128];    // [key][feat]        8 KB
  __shared__ unsigned short lds_vt[128][32];   // [d][key] (transp.) 8 KB
  __shared__ unsigned short lds_p[4][16][32];  // per-wave P patch   4 KB

  const int t = threadIdx.x;
  const int lane = t & 31;
  const int wid = t >> 5;                       // 0..3
  const int blk = blockIdx.x;                   // S/64 blocks
  const int hh = blockIdx.y;                    // head
  const int qbase = blk * 64 + wid * 16;
  const float scale = 0.08838834764831845f;     // 1/sqrt(128)

  // ---- load Q fragments once: 4 chunks of K=32 over D=128
  v16bf qf[4];
  {
    int qrow = qbase + (lane & 15);
#pragma unroll
    for (int c = 0; c < 4; ++c) {
      U8x4 qu;
#pragma unroll
      for (int v = 0; v < 8; ++v) {
        int k0 = 32 * c + a_k0(v, lane);
        qu.u[v] = *(const unsigned int*)&Qb[((size_t)qrow * H_NUM + hh) * D_DIM + k0];
      }
      qf[c] = __builtin_bit_cast(v16bf, qu);
    }
  }

  float rowmax[8], rowsum[8];
#pragma unroll
  for (int r = 0; r < 8; ++r) { rowmax[r] = -3.0e38f; rowsum[r] = 0.0f; }
  v8f acc[8] = {};                              // 16 x 128 output accumulator

  const int ntiles = 2 * blk + 2;               // causal bound for q = blk*64+63
  for (int j = 0; j < ntiles; ++j) {
    const int kt = j * 32;

    // ---- async stage K tile row-major (4 b128 per thread, no VGPR round trip)
#pragma unroll
    for (int i = 0; i < 4; ++i) {
      int c = t * 4 + i;                        // 0..511
      int krow = c >> 4;                        // 0..31 (key within tile)
      int dc8 = (c & 15) * 8;                   // 0..120 (feature chunk)
      async_copy_b128(&Kb[((size_t)(kt + krow) * H_NUM + hh) * D_DIM + dc8],
                      &lds_k[krow][dc8]);
    }
    // ---- stage V tile transposed: key pairs packed -> ds_store_b32
#pragma unroll
    for (int i = 0; i < 2; ++i) {
      int c = t * 2 + i;                        // 0..255
      int kp = (c >> 4) * 2;                    // even key 0..30
      int dc8 = (c & 15) * 8;                   // 0..120
      uint4 g0 = *(const uint4*)&Vb[((size_t)(kt + kp) * H_NUM + hh) * D_DIM + dc8];
      uint4 g1 = *(const uint4*)&Vb[((size_t)(kt + kp + 1) * H_NUM + hh) * D_DIM + dc8];
      unsigned int w0[4] = {g0.x, g0.y, g0.z, g0.w};
      unsigned int w1[4] = {g1.x, g1.y, g1.z, g1.w};
#pragma unroll
      for (int e = 0; e < 8; ++e) {
        unsigned lo = (w0[e >> 1] >> ((e & 1) * 16)) & 0xffffu;
        unsigned hi = (w1[e >> 1] >> ((e & 1) * 16)) & 0xffffu;
        *(unsigned int*)&lds_vt[dc8 + e][kp] = lo | (hi << 16);
      }
    }
    // ---- prefetch next tile
    if (j + 1 < ntiles) {
      int krow = (t * 4) >> 4;
      int dc8 = ((t * 4) & 15) * 8;
      __builtin_prefetch(&Kb[((size_t)(kt + 32 + krow) * H_NUM + hh) * D_DIM + dc8], 0, 3);
      __builtin_prefetch(&Vb[((size_t)(kt + 32 + krow) * H_NUM + hh) * D_DIM + dc8], 0, 3);
    }
    wait_async();
    __syncthreads();

    const bool active = (kt <= qbase + 15);     // wave-uniform
    if (active) {
      // ---- scores for two 16-key halves of this tile
      v8f sv[2];
#pragma unroll
      for (int t2 = 0; t2 < 2; ++t2) {
        v8f s = {};
#pragma unroll
        for (int c = 0; c < 4; ++c) {
          int krow = 16 * t2 + (lane & 15);
          int foff = 32 * c + ((lane & 16) ? 16 : 0);
          v16bf kfrag = *(const v16bf*)&lds_k[krow][foff];
          s = __builtin_amdgcn_wmma_f32_16x16x32_bf16(
              false, qf[c], false, kfrag, (short)0, s, false, false);
        }
        sv[t2] = s;
      }

      // ---- online softmax (row r lives across one 16-lane half)
#pragma unroll
      for (int r = 0; r < 8; ++r) {
        int qi = qbase + r + ((lane & 16) ? 8 : 0);
        int ki0 = kt + (lane & 15);
        float v0 = sv[0][r] * scale; if (ki0 > qi)      v0 -= 1.0e9f;
        float v1 = sv[1][r] * scale; if (ki0 + 16 > qi) v1 -= 1.0e9f;
        float nm = fmaxf(rowmax[r], redmax16(fmaxf(v0, v1)));
        float corr = __expf(rowmax[r] - nm);
        float p0 = __expf(v0 - nm);
        float p1 = __expf(v1 - nm);
        rowsum[r] = rowsum[r] * corr + redsum16(p0 + p1);
        rowmax[r] = nm;
        int m = r + ((lane & 16) ? 8 : 0);
        lds_p[wid][m][(lane & 15)]      = f2bf(p0);
        lds_p[wid][m][16 + (lane & 15)] = f2bf(p1);
#pragma unroll
        for (int d = 0; d < 8; ++d) acc[d][r] *= corr;
      }

      // ---- P (16x32) as A-fragment (LDS ops are in-order within a wave)
      U8x4 pu;
#pragma unroll
      for (int v = 0; v < 8; ++v)
        pu.u[v] = *(const unsigned int*)&lds_p[wid][lane & 15][a_k0(v, lane)];
      v16bf pfrag = __builtin_bit_cast(v16bf, pu);

      // ---- acc += P * V
#pragma unroll
      for (int d = 0; d < 8; ++d) {
        int dcol = d * 16 + (lane & 15);
        int koff = (lane & 16) ? 16 : 0;
        v16bf vfrag = *(const v16bf*)&lds_vt[dcol][koff];
        acc[d] = __builtin_amdgcn_wmma_f32_16x16x32_bf16(
            false, pfrag, false, vfrag, (short)0, acc[d], false, false);
      }
    }
    __syncthreads();
  }

  // ---- normalize and store bf16 [S][H*D]
#pragma unroll
  for (int d = 0; d < 8; ++d) {
#pragma unroll
    for (int r = 0; r < 8; ++r) {
      int m = r + ((lane & 16) ? 8 : 0);
      int q = qbase + m;
      float o = acc[d][r] / rowsum[r];
      Ob[((size_t)q * H_NUM + hh) * D_DIM + d * 16 + (lane & 15)] = f2bf(o);
    }
  }
}

// ---------------------------------------------------------------------------
// Host-side orchestration (all on `stream`, graph-capture safe).
// Inputs: x, Wq, Wk, Wv, Wo, mask (unused; causal mask applied in-kernel),
//         start_pos (unused; reference uses arange(seqlen)).
// Workspace layout (bf16 buffers), total 112 MB:
//   xb[S*E], wbuf[E*HD] (reused per weight), qb/kb/vb[S*HD], ab[S*HD]
// ---------------------------------------------------------------------------
extern "C" void kernel_launch(void* const* d_in, const int* in_sizes, int n_in,
                              void* d_out, int out_size, void* d_ws, size_t ws_size,
                              hipStream_t stream) {
  const float* x  = (const float*)d_in[0];
  const float* Wq = (const float*)d_in[1];
  const float* Wk = (const float*)d_in[2];
  const float* Wv = (const float*)d_in[3];
  const float* Wo = (const float*)d_in[4];
  (void)in_sizes; (void)n_in; (void)out_size; (void)ws_size;

  unsigned char* ws = (unsigned char*)d_ws;
  size_t off = 0;
  unsigned short* xb   = (unsigned short*)(ws + off); off += (size_t)S_LEN * E_DIM * 2;
  unsigned short* wbuf = (unsigned short*)(ws + off); off += (size_t)E_DIM * HD * 2;
  unsigned short* qb   = (unsigned short*)(ws + off); off += (size_t)S_LEN * HD * 2;
  unsigned short* kb   = (unsigned short*)(ws + off); off += (size_t)S_LEN * HD * 2;
  unsigned short* vb   = (unsigned short*)(ws + off); off += (size_t)S_LEN * HD * 2;
  unsigned short* ab   = (unsigned short*)(ws + off); off += (size_t)S_LEN * HD * 2;

  const int nx4 = (S_LEN * E_DIM) / 4;
  const int nw4 = (E_DIM * HD) / 4;
  dim3 cb(256);
  dim3 gemm_grid(HD / 128, S_LEN / 64);   // N/128, M/64

  // x -> bf16
  cvt_f32_bf16<<<dim3((nx4 + 255) / 256), cb, 0, stream>>>(x, xb, nx4);

  // Q = x @ Wq
  cvt_f32_bf16<<<dim3((nw4 + 255) / 256), cb, 0, stream>>>(Wq, wbuf, nw4);
  gemm_wmma_bf16<true><<<gemm_grid, dim3(256), 0, stream>>>(xb, wbuf, qb, S_LEN, HD, E_DIM);
  // K = x @ Wk
  cvt_f32_bf16<<<dim3((nw4 + 255) / 256), cb, 0, stream>>>(Wk, wbuf, nw4);
  gemm_wmma_bf16<true><<<gemm_grid, dim3(256), 0, stream>>>(xb, wbuf, kb, S_LEN, HD, E_DIM);
  // V = x @ Wv
  cvt_f32_bf16<<<dim3((nw4 + 255) / 256), cb, 0, stream>>>(Wv, wbuf, nw4);
  gemm_wmma_bf16<true><<<gemm_grid, dim3(256), 0, stream>>>(xb, wbuf, vb, S_LEN, HD, E_DIM);

  // RoPE on Q, K
  const int nrope = S_LEN * H_NUM * (D_DIM / 2);
  rope_bf16<<<dim3((nrope + 255) / 256), cb, 0, stream>>>(qb);
  rope_bf16<<<dim3((nrope + 255) / 256), cb, 0, stream>>>(kb);

  // Flash attention
  attn_flash_wmma<<<dim3(S_LEN / 64, H_NUM), dim3(128), 0, stream>>>(qb, kb, vb, ab);

  // out = attn @ Wo  (fp32 straight into d_out)
  cvt_f32_bf16<<<dim3((nw4 + 255) / 256), cb, 0, stream>>>(Wo, wbuf, nw4);
  gemm_wmma_bf16<false><<<dim3(E_DIM / 128, S_LEN / 64), dim3(256), 0, stream>>>(
      ab, wbuf, (float*)d_out, S_LEN, E_DIM, HD);
}
